// MultiHeadAttention_14834817040461
// MI455X (gfx1250) — matrix operand order repaired and checked by reading, not compiled, not run on hardware
//
#include <hip/hip_runtime.h>

// ---------------------------------------------------------------------------
// MI455X (gfx1250) multi-head attention, wave32 + WMMA f16->f32.
// Pipeline: f16 convert -> transposed f16 activations -> WMMA QKV projection
//           (V stored transposed) -> flash-style WMMA attention -> WMMA out
//           projection. GEMMs use ping-pong register double buffering
//           (peeled prologue/epilogue: no buffer-rotate copies).
// ---------------------------------------------------------------------------

typedef __attribute__((ext_vector_type(16))) _Float16 v16h;
typedef __attribute__((ext_vector_type(8)))  _Float16 v8h;
typedef __attribute__((ext_vector_type(8)))  float    v8f;

constexpr int Bc  = 2;
constexpr int Dc  = 1024;
constexpr int Nc  = 2048;
constexpr int Hc  = 16;
constexpr int DHc = 64;   // head dim

// D = A*B + C, A/B f16 16x32 / 32x16, C/D f32 16x16
__device__ __forceinline__ v8f wmma_f16(v16h a, v16h b, v8f c) {
  return __builtin_amdgcn_wmma_f32_16x16x32_f16(false, a, false, b,
                                                (short)0, c, false, false);
}

// A-matrix fragment (16x32 f16, row-major source, row stride ld halves).
// Lane holds row (lane%16); halves 0..15 = K 16*(lane/16)..+15
// -> two contiguous 16B loads.
__device__ __forceinline__ v16h load_a_frag(const _Float16* __restrict__ base,
                                            int ld, int row, int k0, int h2) {
  const _Float16* p = base + (size_t)row * ld + k0 + 16 * h2;
  v8h lo = *(const v8h*)p;
  v8h hi = *(const v8h*)(p + 8);
  v16h r;
#pragma unroll
  for (int i = 0; i < 8; ++i) { r[i] = lo[i]; r[i + 8] = hi[i]; }
  return r;
}

// ---------------------------------------------------------------------------
// 1) f32 -> f16 elementwise (weights)
// ---------------------------------------------------------------------------
__global__ __launch_bounds__(256) void f32_to_f16_kernel(
    const float* __restrict__ src, _Float16* __restrict__ dst, int n) {
  int i = blockIdx.x * blockDim.x + threadIdx.x;
  if (i < n) dst[i] = (_Float16)src[i];
}

// ---------------------------------------------------------------------------
// 2) query [B,D,N] f32 -> XT [B,N,D] f16, LDS tiled transpose
// ---------------------------------------------------------------------------
__global__ __launch_bounds__(256) void transpose_to_f16_kernel(
    const float* __restrict__ src, _Float16* __restrict__ dst) {
  __shared__ float tile[32][33];
  const int b  = blockIdx.z;
  const int d0 = blockIdx.y * 32;
  const int n0 = blockIdx.x * 32;
  const int tx = threadIdx.x;            // 0..31
  const int ty = threadIdx.y;            // 0..7
#pragma unroll
  for (int r = ty; r < 32; r += 8)
    tile[r][tx] = src[((size_t)b * Dc + d0 + r) * Nc + n0 + tx];
  __syncthreads();
#pragma unroll
  for (int r = ty; r < 32; r += 8)
    dst[((size_t)b * Nc + n0 + r) * Dc + d0 + tx] = (_Float16)tile[tx][r];
}

// ---------------------------------------------------------------------------
// Shared GEMM core: acc[j] (16e x 4*16n) = W16[e,:] x act[b,n,:]^T.
// Ping-pong register double buffering, peeled prologue/epilogue.
// ---------------------------------------------------------------------------
__device__ __forceinline__ void gemm_core(const _Float16* __restrict__ W16,
                                          const _Float16* __restrict__ act,
                                          int b, int e0, int n0, int l16,
                                          int h2, v8f acc[4]) {
  const _Float16* arow = W16 + (size_t)(e0 + l16) * Dc + 16 * h2;
  const _Float16* xrow[4];
#pragma unroll
  for (int j = 0; j < 4; ++j)
    xrow[j] = act + ((size_t)b * Nc + n0 + j * 16 + l16) * Dc + 16 * h2;

  v16h a0, a1, b0[4], b1[4];

  // prologue: buffer 0 <- k = 0
  {
    v8h lo = *(const v8h*)(arow);
    v8h hi = *(const v8h*)(arow + 8);
#pragma unroll
    for (int i = 0; i < 8; ++i) { a0[i] = lo[i]; a0[i + 8] = hi[i]; }
#pragma unroll
    for (int j = 0; j < 4; ++j) b0[j] = *(const v16h*)(xrow[j]);
  }

#pragma unroll 1
  for (int k0 = 0; k0 < Dc - 64; k0 += 64) {
    // buffer 1 <- k0+32
    {
      v8h lo = *(const v8h*)(arow + k0 + 32);
      v8h hi = *(const v8h*)(arow + k0 + 40);
#pragma unroll
      for (int i = 0; i < 8; ++i) { a1[i] = lo[i]; a1[i + 8] = hi[i]; }
#pragma unroll
      for (int j = 0; j < 4; ++j) b1[j] = *(const v16h*)(xrow[j] + k0 + 32);
    }
#pragma unroll
    for (int j = 0; j < 4; ++j) acc[j] = wmma_f16(a0, b0[j], acc[j]);
    // buffer 0 <- k0+64
    {
      v8h lo = *(const v8h*)(arow + k0 + 64);
      v8h hi = *(const v8h*)(arow + k0 + 72);
#pragma unroll
      for (int i = 0; i < 8; ++i) { a0[i] = lo[i]; a0[i + 8] = hi[i]; }
#pragma unroll
      for (int j = 0; j < 4; ++j) b0[j] = *(const v16h*)(xrow[j] + k0 + 64);
    }
#pragma unroll
    for (int j = 0; j < 4; ++j) acc[j] = wmma_f16(a1, b1[j], acc[j]);
  }

  // epilogue: buffer 0 holds k = Dc-64; load buffer 1 <- Dc-32
  {
    v8h lo = *(const v8h*)(arow + Dc - 32);
    v8h hi = *(const v8h*)(arow + Dc - 24);
#pragma unroll
    for (int i = 0; i < 8; ++i) { a1[i] = lo[i]; a1[i + 8] = hi[i]; }
#pragma unroll
    for (int j = 0; j < 4; ++j) b1[j] = *(const v16h*)(xrow[j] + Dc - 32);
  }
#pragma unroll
  for (int j = 0; j < 4; ++j) acc[j] = wmma_f16(a0, b0[j], acc[j]);
#pragma unroll
  for (int j = 0; j < 4; ++j) acc[j] = wmma_f16(a1, b1[j], acc[j]);
}

// ---------------------------------------------------------------------------
// 3a) Q/K projection: store f16 head-split [B,H,N,DH] (n-major rows).
// ---------------------------------------------------------------------------
__global__ __launch_bounds__(128) void proj_qk_kernel(
    const _Float16* __restrict__ W16, const float* __restrict__ bias,
    const _Float16* __restrict__ XT, _Float16* __restrict__ dst) {
  const int lane = threadIdx.x & 31;
  const int wave = threadIdx.x >> 5;
  const int l16  = lane & 15;
  const int h2   = lane >> 4;
  const int b    = blockIdx.z;
  const int e0   = blockIdx.y * 64 + wave * 16;
  const int n0   = blockIdx.x * 64;

  v8f acc[4] = {};
  gemm_core(W16, XT, b, e0, n0, l16, h2, acc);

  // C layout: (vgpr v, lane) -> e = e0 + 8*h2 + v, n = n0 + 16j + l16.
  const int ebase = e0 + 8 * h2;
  const int hd    = ebase >> 6;
  const int dh0   = ebase & 63;
#pragma unroll
  for (int j = 0; j < 4; ++j) {
    const int n = n0 + j * 16 + l16;
    v8h pk;
#pragma unroll
    for (int v = 0; v < 8; ++v)
      pk[v] = (_Float16)(acc[j][v] + bias[ebase + v]);
    _Float16* op = dst + (((size_t)b * Hc + hd) * Nc + n) * DHc + dh0;
    *(v8h*)op = pk;   // packed 16B store (free transpose from C layout)
  }
}

// ---------------------------------------------------------------------------
// 3b) V projection stored TRANSPOSED: VT f16 [B*H, DH, N] (natural C layout)
//     so attention PV B-fragments become contiguous 32B loads.
// ---------------------------------------------------------------------------
__global__ __launch_bounds__(128) void proj_vt_kernel(
    const _Float16* __restrict__ W16, const float* __restrict__ bias,
    const _Float16* __restrict__ XT, _Float16* __restrict__ dst) {
  const int lane = threadIdx.x & 31;
  const int wave = threadIdx.x >> 5;
  const int l16  = lane & 15;
  const int h2   = lane >> 4;
  const int b    = blockIdx.z;
  const int e0   = blockIdx.y * 64 + wave * 16;
  const int n0   = blockIdx.x * 64;

  v8f acc[4] = {};
  gemm_core(W16, XT, b, e0, n0, l16, h2, acc);

  const int ebase = e0 + 8 * h2;
  const int hd    = ebase >> 6;
  const int dh0   = ebase & 63;       // + v stays within one head (<=63)
  _Float16* vt = dst + ((size_t)(b * Hc + hd) * DHc + dh0) * Nc;
#pragma unroll
  for (int j = 0; j < 4; ++j) {
    const int n = n0 + j * 16 + l16;
#pragma unroll
    for (int v = 0; v < 8; ++v)
      vt[(size_t)v * Nc + n] = (_Float16)(acc[j][v] + bias[ebase + v]);
  }
}

// ---------------------------------------------------------------------------
// 4) Flash-style attention. One wave per 16 query rows.
//    Q,K f16 [B*H,N,DH]; VT f16 [B*H,DH,N]; mask [B,N] bytes; Y f16 [B,N,D].
// ---------------------------------------------------------------------------
__global__ __launch_bounds__(128) void attn_kernel(
    const _Float16* __restrict__ Q, const _Float16* __restrict__ K,
    const _Float16* __restrict__ VT, const unsigned char* __restrict__ mask,
    _Float16* __restrict__ Y) {
  __shared__ _Float16 Pl[4][16 * 32];   // per-wave P tile (16q x 32k) f16

  const int lane = threadIdx.x & 31;
  const int wave = threadIdx.x >> 5;
  const int l16  = lane & 15;
  const int h2   = lane >> 4;
  const int bh   = blockIdx.y;                 // 0..B*H-1
  const int b    = bh / Hc;
  const int hd   = bh % Hc;
  const int q0   = blockIdx.x * 64 + wave * 16;
  const size_t kbase  = (size_t)bh * Nc * DHc;
  const size_t vtbase = (size_t)bh * DHc * Nc;
  const float scale = 0.125f;                  // 1/sqrt(64)

  // Q A-fragments: rows q0..q0+15, dh 0..31 and 32..63
  const v16h aq0 = load_a_frag(Q + kbase, DHc, q0 + l16, 0,  h2);
  const v16h aq1 = load_a_frag(Q + kbase, DHc, q0 + l16, 32, h2);

  v8f o[4] = {};                // 16q x 64dh accumulator (4 C tiles)
  float mrow[8], lrow[8];
#pragma unroll
  for (int v = 0; v < 8; ++v) { mrow[v] = -__builtin_inff(); lrow[v] = 0.0f; }

  _Float16* pb = Pl[wave];

#pragma unroll 1
  for (int k0 = 0; k0 < Nc; k0 += 32) {
    // ---- issue all global loads for this step up front ----
    const _Float16* kp0 = K + kbase + (size_t)(k0 + l16) * DHc + 16 * h2;
    const _Float16* kp1 = K + kbase + (size_t)(k0 + 16 + l16) * DHc + 16 * h2;
    v16h kf0 = *(const v16h*)kp0;
    v16h kf1 = *(const v16h*)(kp0 + 32);
    v16h kf2 = *(const v16h*)kp1;
    v16h kf3 = *(const v16h*)(kp1 + 32);
    v16h vf[4];
#pragma unroll
    for (int t = 0; t < 4; ++t) {
      const _Float16* vp =
          VT + vtbase + (size_t)(t * 16 + l16) * Nc + k0 + 16 * h2;
      vf[t] = *(const v16h*)vp;      // contiguous: stays in flight thru softmax
    }
    if (k0 + 32 < Nc) {              // prefetch next key tile (L2-resident)
      __builtin_prefetch(K + kbase + (size_t)(k0 + 32 + l16) * DHc, 0, 3);
      __builtin_prefetch(VT + vtbase + (size_t)l16 * Nc + k0 + 32, 0, 3);
    }

    // ---- S = (Q*scale) K^T : two 16x16 tiles, each 2 chained WMMAs ----
    v8f s0 = {}, s1 = {};
    s0 = wmma_f16(aq0, kf0, s0);
    s0 = wmma_f16(aq1, kf1, s0);
    s1 = wmma_f16(aq0, kf2, s1);
    s1 = wmma_f16(aq1, kf3, s1);

    const bool ms0 = mask[(size_t)b * Nc + k0 + l16] != 0;
    const bool ms1 = mask[(size_t)b * Nc + k0 + 16 + l16] != 0;

    // ---- online softmax: each accum reg v is one q-row (cols across lanes)
    float p0[8], p1[8];
#pragma unroll
    for (int v = 0; v < 8; ++v) {
      float sv0 = ms0 ? s0[v] * scale : -__builtin_inff();
      float sv1 = ms1 ? s1[v] * scale : -__builtin_inff();
      float cm  = fmaxf(sv0, sv1);
      cm = fmaxf(cm, __shfl_xor(cm, 1, 32));
      cm = fmaxf(cm, __shfl_xor(cm, 2, 32));
      cm = fmaxf(cm, __shfl_xor(cm, 4, 32));
      cm = fmaxf(cm, __shfl_xor(cm, 8, 32));
      const float mn = fmaxf(mrow[v], cm);
      const float alpha =
          (mn == -__builtin_inff()) ? 1.0f : __expf(mrow[v] - mn);
      p0[v] = ms0 ? __expf(sv0 - mn) : 0.0f;
      p1[v] = ms1 ? __expf(sv1 - mn) : 0.0f;
      float rs = p0[v] + p1[v];
      rs += __shfl_xor(rs, 1, 32);
      rs += __shfl_xor(rs, 2, 32);
      rs += __shfl_xor(rs, 4, 32);
      rs += __shfl_xor(rs, 8, 32);
      lrow[v] = lrow[v] * alpha + rs;
      mrow[v] = mn;
#pragma unroll
      for (int t = 0; t < 4; ++t) o[t][v] *= alpha;
    }

    // ---- P: C layout -> A layout via per-wave LDS buffer ----
#pragma unroll
    for (int v = 0; v < 8; ++v) {
      const int r = v + 8 * h2;
      pb[r * 32 + l16]      = (_Float16)p0[v];
      pb[r * 32 + 16 + l16] = (_Float16)p1[v];
    }
    __syncthreads();   // intra-wave ds ordering; uniform across block
    v16h aP;
    {
      const _Float16* p = pb + l16 * 32 + 16 * h2;
      v8h lo = *(const v8h*)p;
      v8h hi = *(const v8h*)(p + 8);
#pragma unroll
      for (int i = 0; i < 8; ++i) { aP[i] = lo[i]; aP[i + 8] = hi[i]; }
    }

    // ---- O += P (16x32) x V (32xDH) : 4 WMMAs over dh tiles ----
#pragma unroll
    for (int t = 0; t < 4; ++t) o[t] = wmma_f16(aP, vf[t], o[t]);
  }

  // ---- epilogue: normalize and store to Y [B, N, D] (d = head*64+dh) ----
#pragma unroll
  for (int v = 0; v < 8; ++v) {
    const float inv = (lrow[v] > 0.0f) ? 1.0f / lrow[v] : 0.0f;
    const int q = q0 + v + 8 * h2;
#pragma unroll
    for (int t = 0; t < 4; ++t) {
      const int d = hd * DHc + t * 16 + l16;
      Y[((size_t)b * Nc + q) * Dc + d] = (_Float16)(o[t][v] * inv);
    }
  }
}

// ---------------------------------------------------------------------------
// 5) Output projection: out[b,e,n] = sum_d Wo16[e,d]*Y[b,n,d] + bo[e], f32
// ---------------------------------------------------------------------------
__global__ __launch_bounds__(128) void out_proj_kernel(
    const _Float16* __restrict__ W16, const float* __restrict__ bias,
    const _Float16* __restrict__ Y, float* __restrict__ out) {
  const int lane = threadIdx.x & 31;
  const int wave = threadIdx.x >> 5;
  const int l16  = lane & 15;
  const int h2   = lane >> 4;
  const int b    = blockIdx.z;
  const int e0   = blockIdx.y * 64 + wave * 16;
  const int n0   = blockIdx.x * 64;

  v8f acc[4] = {};
  gemm_core(W16, Y, b, e0, n0, l16, h2, acc);

  const int ebase = e0 + 8 * h2;
#pragma unroll
  for (int j = 0; j < 4; ++j) {
    const int n = n0 + j * 16 + l16;
#pragma unroll
    for (int v = 0; v < 8; ++v) {
      const int e = ebase + v;
      out[((size_t)b * Dc + e) * Nc + n] = acc[j][v] + bias[e];
    }
  }
}

// ---------------------------------------------------------------------------
// launch
// ---------------------------------------------------------------------------
extern "C" void kernel_launch(void* const* d_in, const int* in_sizes, int n_in,
                              void* d_out, int out_size, void* d_ws,
                              size_t ws_size, hipStream_t stream) {
  (void)in_sizes; (void)n_in; (void)out_size; (void)ws_size;

  const float* query = (const float*)d_in[0];
  const float* Wq    = (const float*)d_in[1];
  const float* bq    = (const float*)d_in[2];
  const float* Wk    = (const float*)d_in[3];
  const float* bk    = (const float*)d_in[4];
  const float* Wv    = (const float*)d_in[5];
  const float* bv    = (const float*)d_in[6];
  const float* Wo    = (const float*)d_in[7];
  const float* bo    = (const float*)d_in[8];
  const unsigned char* mask = (const unsigned char*)d_in[9];  // bool [B,1,N]
  float* out = (float*)d_out;

  // workspace layout (48 MB total)
  const size_t MB = 1u << 20;
  char* ws = (char*)d_ws;
  _Float16* Wq16 = (_Float16*)(ws + 0 * MB);
  _Float16* Wk16 = (_Float16*)(ws + 2 * MB);
  _Float16* Wv16 = (_Float16*)(ws + 4 * MB);
  _Float16* Wo16 = (_Float16*)(ws + 6 * MB);
  _Float16* XT   = (_Float16*)(ws + 8 * MB);    // [B,N,D] f16
  _Float16* Q16  = (_Float16*)(ws + 16 * MB);   // [B*H,N,DH] f16
  _Float16* K16  = (_Float16*)(ws + 24 * MB);   // [B*H,N,DH] f16
  _Float16* V16T = (_Float16*)(ws + 32 * MB);   // [B*H,DH,N] f16 (transposed)
  _Float16* Y16  = (_Float16*)(ws + 40 * MB);   // [B,N,D] f16

  // 1) weight conversion
  const int nW = Dc * Dc;
  dim3 cgrid((nW + 255) / 256);
  f32_to_f16_kernel<<<cgrid, 256, 0, stream>>>(Wq, Wq16, nW);
  f32_to_f16_kernel<<<cgrid, 256, 0, stream>>>(Wk, Wk16, nW);
  f32_to_f16_kernel<<<cgrid, 256, 0, stream>>>(Wv, Wv16, nW);
  f32_to_f16_kernel<<<cgrid, 256, 0, stream>>>(Wo, Wo16, nW);

  // 2) query transpose [B,D,N] -> [B,N,D] f16
  transpose_to_f16_kernel<<<dim3(Nc / 32, Dc / 32, Bc), dim3(32, 8), 0,
                            stream>>>(query, XT);

  // 3) QKV projections (V transposed)
  dim3 pgrid(Nc / 64, Dc / 64, Bc);
  proj_qk_kernel<<<pgrid, 128, 0, stream>>>(Wq16, bq, XT, Q16);
  proj_qk_kernel<<<pgrid, 128, 0, stream>>>(Wk16, bk, XT, K16);
  proj_vt_kernel<<<pgrid, 128, 0, stream>>>(Wv16, bv, XT, V16T);

  // 4) attention
  attn_kernel<<<dim3(Nc / 64, Bc * Hc), 128, 0, stream>>>(Q16, K16, V16T,
                                                          mask, Y16);

  // 5) output projection
  out_proj_kernel<<<pgrid, 128, 0, stream>>>(Wo16, bo, Y16, out);
}
